// ProcessorNetwork_23665269801224
// MI455X (gfx1250) — matrix operand order repaired and checked by reading, not compiled, not run on hardware
//
#include <hip/hip_runtime.h>
#include <hip/hip_bf16.h>

// ---------------------------------------------------------------------------
// Problem constants (from reference): B=8, N=2048, D=128
// ---------------------------------------------------------------------------
constexpr int Bn = 8;
constexpr int Nn = 2048;
constexpr int Dn = 128;

constexpr int JT   = 64;    // j-tile (output rows) per workgroup
constexpr int KT   = 32;    // K step (bf16 WMMA K)
constexpr int KPAD = 40;    // LDS row stride (elems): 80B = 64B row + 16B TDM pad
constexpr int DPAD = 136;   // LDS row stride (elems) for 128-wide rows (272B, 16B aligned)

typedef __attribute__((ext_vector_type(16))) __bf16 v16bf;
typedef __attribute__((ext_vector_type(8)))  float  v8f;
typedef __attribute__((ext_vector_type(4)))  unsigned int u32x4;
typedef __attribute__((ext_vector_type(8)))  int          i32x8;
typedef __attribute__((ext_vector_type(4)))  int          i32x4;

union Frag {
    uint4 q[2];
    v16bf v;
};

// ---------------------------------------------------------------------------
// Kernel 1: invDeg[b*N+i] = 1 / sum_j adj[b,i,j].  One wave32 per row.
// ---------------------------------------------------------------------------
__global__ __launch_bounds__(256) void deg_kernel(const float* __restrict__ adj,
                                                  float* __restrict__ invDeg) {
    const int w    = threadIdx.x >> 5;
    const int lane = threadIdx.x & 31;
    const int row  = blockIdx.x * 8 + w;              // 0 .. B*N-1
    const float4* p = (const float4*)(adj + (size_t)row * Nn);
    float s = 0.f;
#pragma unroll
    for (int it = 0; it < Nn / 4 / 32; ++it) {        // 16 float4 per lane
        float4 v = p[lane + 32 * it];
        s += v.x + v.y + v.z + v.w;
    }
#pragma unroll
    for (int off = 16; off > 0; off >>= 1) s += __shfl_down(s, off, 32);
    if (lane == 0) invDeg[row] = 1.0f / s;
}

// ---------------------------------------------------------------------------
// Kernel 2: convert W (D x 2D f32, row-major) to bf16 row-major (stride 256).
// ---------------------------------------------------------------------------
__global__ __launch_bounds__(256) void prepw_kernel(const float* __restrict__ Wf,
                                                    __bf16* __restrict__ Wbf) {
    int t = blockIdx.x * 256 + threadIdx.x;           // 0 .. 8191 float4's
    float4 v = ((const float4*)Wf)[t];
    __bf16* d = Wbf + (size_t)t * 4;
    d[0] = (__bf16)v.x; d[1] = (__bf16)v.y; d[2] = (__bf16)v.z; d[3] = (__bf16)v.w;
}

// ---------------------------------------------------------------------------
// Kernel 3: embT[b][d][i] = bf16(emb[b][i][d])  (one-time LDS-tiled transpose)
// Grid: (N/64, B), 256 threads.
// ---------------------------------------------------------------------------
__global__ __launch_bounds__(256) void prep_embT(const float* __restrict__ emb,
                                                 __bf16* __restrict__ embT) {
    __shared__ __bf16 T[64][DPAD];                    // [i_local][d]
    const int b   = blockIdx.y;
    const int i0  = blockIdx.x * 64;
    const int tid = threadIdx.x;

    const float4* src = (const float4*)(emb + ((size_t)b * Nn + i0) * Dn);
#pragma unroll
    for (int r = 0; r < 8; ++r) {                     // 64x128 f32 = 2048 float4
        int q  = tid + 256 * r;
        int i  = q >> 5;
        int dq = (q & 31) * 4;
        float4 v = src[q];
        __bf16* dst = &T[i][dq];
        dst[0] = (__bf16)v.x; dst[1] = (__bf16)v.y; dst[2] = (__bf16)v.z; dst[3] = (__bf16)v.w;
    }
    __syncthreads();

    const int d  = tid >> 1;                          // 0..127
    const int ih = (tid & 1) * 32;                    // i-half
    __bf16 tmp[32];
#pragma unroll
    for (int ii = 0; ii < 32; ++ii) tmp[ii] = T[ih + ii][d];
    __bf16* dst = embT + ((size_t)b * Dn + d) * Nn + i0 + ih;  // 64B aligned
    const uint4* s4 = (const uint4*)tmp;
    uint4* d4 = (uint4*)dst;
#pragma unroll
    for (int c = 0; c < 4; ++c) d4[c] = s4[c];
}

// ---------------------------------------------------------------------------
// Kernel 4: fused  S = cost^T E  ->  out = relu([E | S/deg] W^T + b)
// Grid: (N/JT, B), 256 threads (8 waves).
// Wave w: j-rows [ (w&3)*16 , +16 ),  d-half (w>>2)*64.
// B-tiles (emb^T) are staged by the Tensor Data Mover (one descriptor per
// K-step, issued by wave 0, TDM pad feature reproduces the KPAD bank skew).
// ---------------------------------------------------------------------------
__global__ __launch_bounds__(256) void fused_kernel(const float* __restrict__ emb,
                                                    const float* __restrict__ cost,
                                                    const float* __restrict__ bias,
                                                    const float* __restrict__ invDeg,
                                                    const __bf16* __restrict__ Wbf,
                                                    const __bf16* __restrict__ embT,
                                                    float* __restrict__ out) {
    __shared__ __bf16 Albuf[2][JT][KPAD];   // cost^T tile [j][k]
    __shared__ __bf16 Blbuf[2][Dn][KPAD];   // emb^T  tile [d][k]  (filled by TDM)
    __shared__ __bf16 Ej[JT][DPAD];         // emb j-rows  [j][d]
    __shared__ __bf16 Sc[JT][DPAD];         // scaled S    [j][d]

    const int b    = blockIdx.y;
    const int j0   = blockIdx.x * JT;
    const int tid  = threadIdx.x;
    const int lane = tid & 31;
    const int w    = tid >> 5;
    const int m    = lane & 15;
    const int half = lane >> 4;
    const int jw   = (w & 3) * 16;
    const int dh   = w >> 2;

    const float*  embB  = emb  + (size_t)b * Nn * Dn;
    const float*  costB = cost + (size_t)b * Nn * Nn;
    const __bf16* embTB = embT + (size_t)b * Dn * Nn;

    // ---- stage E rows for this j-tile (A-matrix of epilogue GEMM) ----
    {
        const float4* src = (const float4*)(embB + (size_t)j0 * Dn);  // 2048 float4
#pragma unroll
        for (int r = 0; r < 8; ++r) {
            int q  = tid + 256 * r;
            int j  = q >> 5;
            int dq = (q & 31) * 4;
            float4 v = src[q];
            __bf16* dst = &Ej[j][dq];
            dst[0] = (__bf16)v.x; dst[1] = (__bf16)v.y; dst[2] = (__bf16)v.z; dst[3] = (__bf16)v.w;
        }
    }

    // ---- LDS staging for one K-step ----
    auto stage = [&](int it) {
        const int buf = it & 1;
        const int k0  = it * KT;
        // cost tile: f32 load -> bf16 -> transposed store (32 x 64 = 512 float4)
#pragma unroll
        for (int r = 0; r < 2; ++r) {
            int q  = tid + 256 * r;
            int k  = q >> 4;                  // 16 float4 per 64-wide row
            int jq = (q & 15) * 4;
            const float* p = costB + (size_t)(k0 + k) * Nn + j0 + jq;
            if (r == 0 && it + 2 < Nn / KT)   // warm L2 for the tile after next
                __builtin_prefetch(p + 2 * KT * Nn, 0, 1);
            float4 v = *(const float4*)p;
            Albuf[buf][jq + 0][k] = (__bf16)v.x;
            Albuf[buf][jq + 1][k] = (__bf16)v.y;
            Albuf[buf][jq + 2][k] = (__bf16)v.z;
            Albuf[buf][jq + 3][k] = (__bf16)v.w;
        }
        // emb^T tile: single TDM 2D-tile DMA, global bf16 -> LDS (128 x 32 elems)
        if (w == 0) {
            unsigned           ldsa = (unsigned)(uintptr_t)&Blbuf[buf][0][0];
            unsigned long long ga   = (unsigned long long)(uintptr_t)(embTB + k0);
            u32x4 g0;
            g0[0] = 1u;                                              // count=1, user mode
            g0[1] = ldsa;                                            // lds_addr
            g0[2] = (unsigned)ga;                                    // global_addr[31:0]
            g0[3] = (unsigned)((ga >> 32) & 0x01FFFFFFu) | (2u << 30); // addr[56:32], type=2
            i32x8 g1;
            g1[0] = (int)((1u << 16)      // data_size = 2 bytes
                        | (1u << 20)      // pad_enable
                        | (3u << 22)      // pad_interval: every 16 DWORDs (64B row)
                        | (3u << 25));    // pad_amount:   4 DWORDs (16B) -> 80B stride
            g1[1] = (int)((unsigned)(Nn & 0xFFFF) << 16);            // tensor_dim0 lo16
            g1[2] = (int)((Nn >> 16) | ((unsigned)Dn << 16));        // dim0 hi16 | dim1 lo16
            g1[3] = (int)((Dn >> 16) | ((unsigned)KT << 16));        // dim1 hi16 | tile_dim0=32
            g1[4] = (int)Dn;                                         // tile_dim1=128, tile_dim2=0
            g1[5] = (int)Nn;                                         // tensor_dim0_stride lo32
            g1[6] = 0;                                               // stride hi | dim1_stride lo
            g1[7] = 0;
            i32x4 z4 = {};
            i32x8 z8 = {};
            __builtin_amdgcn_tensor_load_to_lds(g0, g1, z4, z4, z8, 0);
        }
    };

    // ---- main GEMM: S-tile accumulation over K = 2048 ----
    v8f acc[4] = {};
    stage(0);
    if (w == 0) __builtin_amdgcn_s_wait_tensorcnt(0);
    __syncthreads();
    for (int it = 0; it < Nn / KT; ++it) {
        if (it + 1 < Nn / KT) stage(it + 1);
        const int buf = it & 1;
        Frag a;                                    // A: 16x32 bf16, rows = j
        a.q[0] = *(const uint4*)&Albuf[buf][jw + m][half * 8];
        a.q[1] = *(const uint4*)&Albuf[buf][jw + m][16 + half * 8];
#pragma unroll
        for (int t = 0; t < 4; ++t) {
            int n = (dh * 4 + t) * 16 + m;         // B: 32x16 bf16, cols = d
            Frag bb;
            bb.q[0] = *(const uint4*)&Blbuf[buf][n][half * 8];
            bb.q[1] = *(const uint4*)&Blbuf[buf][n][16 + half * 8];
            acc[t] = __builtin_amdgcn_wmma_f32_16x16x32_bf16(
                false, a.v, false, bb.v, (short)0, acc[t], false, false);
        }
        if (w == 0) __builtin_amdgcn_s_wait_tensorcnt(0);
        __syncthreads();
    }

    // ---- scale by 1/degree, spill S to LDS as bf16 (A-matrix of epilogue) ----
    float rs[8];
#pragma unroll
    for (int v = 0; v < 8; ++v)
        rs[v] = invDeg[(size_t)b * Nn + j0 + jw + 8 * half + v];
#pragma unroll
    for (int t = 0; t < 4; ++t) {
        int n = (dh * 4 + t) * 16 + m;
#pragma unroll
        for (int v = 0; v < 8; ++v)
            Sc[jw + 8 * half + v][n] = (__bf16)(acc[t][v] * rs[v]);
    }
    __syncthreads();

    // ---- epilogue GEMM: [E | S/deg] (64x256) x W^T (256x128) + bias, relu ----
    v8f acc2[4];
#pragma unroll
    for (int t = 0; t < 4; ++t) {
        float bv = bias[(dh * 4 + t) * 16 + m];
#pragma unroll
        for (int v = 0; v < 8; ++v) acc2[t][v] = bv;
    }
#pragma unroll
    for (int kk = 0; kk < 8; ++kk) {
        const __bf16* arow = (kk < 4) ? &Ej[jw + m][kk * 32]
                                      : &Sc[jw + m][(kk - 4) * 32];
        Frag a;
        a.q[0] = *(const uint4*)(arow + half * 8);
        a.q[1] = *(const uint4*)(arow + 16 + half * 8);
#pragma unroll
        for (int t = 0; t < 4; ++t) {
            int n = (dh * 4 + t) * 16 + m;
            const __bf16* wrow = Wbf + (size_t)n * 256 + kk * 32 + half * 8;
            Frag bb;
            bb.q[0] = *(const uint4*)(wrow);
            bb.q[1] = *(const uint4*)(wrow + 16);
            acc2[t] = __builtin_amdgcn_wmma_f32_16x16x32_bf16(
                false, a.v, false, bb.v, (short)0, acc2[t], false, false);
        }
    }

    // ---- relu + store ----
    float* outB = out + (size_t)b * Nn * Dn;
#pragma unroll
    for (int t = 0; t < 4; ++t) {
        int n = (dh * 4 + t) * 16 + m;
#pragma unroll
        for (int v = 0; v < 8; ++v) {
            int gj = j0 + jw + 8 * half + v;
            float x = acc2[t][v];
            outB[(size_t)gj * Dn + n] = x > 0.f ? x : 0.f;
        }
    }
}

// ---------------------------------------------------------------------------
// Launch: inputs = {embeddings, adjacency, cost, W, b}
// ws layout: [0,64KB) invDeg f32; [64KB,128KB) W bf16; [128KB, +4MB) embT bf16
// ---------------------------------------------------------------------------
extern "C" void kernel_launch(void* const* d_in, const int* in_sizes, int n_in,
                              void* d_out, int out_size, void* d_ws, size_t ws_size,
                              hipStream_t stream) {
    const float* emb  = (const float*)d_in[0];
    const float* adj  = (const float*)d_in[1];
    const float* cost = (const float*)d_in[2];
    const float* Wf   = (const float*)d_in[3];
    const float* bias = (const float*)d_in[4];
    float* out = (float*)d_out;

    char* ws = (char*)d_ws;
    float*  invDeg = (float*)ws;                                            // 64 KB
    __bf16* Wbf    = (__bf16*)(ws + (size_t)Bn * Nn * sizeof(float));       // 64 KB
    __bf16* embT   = (__bf16*)(ws + (size_t)Bn * Nn * sizeof(float) + (size_t)Dn * 2 * Dn * 2);

    deg_kernel<<<Bn * Nn / 8, 256, 0, stream>>>(adj, invDeg);
    prepw_kernel<<<(Dn * 2 * Dn / 4) / 256, 256, 0, stream>>>(Wf, Wbf);
    prep_embT<<<dim3(Nn / 64, Bn), 256, 0, stream>>>(emb, embT);
    fused_kernel<<<dim3(Nn / JT, Bn), 256, 0, stream>>>(emb, cost, bias, invDeg, Wbf, embT, out);
}
